// Attention_2680059593151
// MI455X (gfx1250) — matrix operand order repaired and checked by reading, not compiled, not run on hardware
//
#include <hip/hip_runtime.h>

// ---------------- problem constants ----------------
#define BATCH 8
#define SEQ   2048
#define FD    512
#define HD    512
#define NROWS (BATCH * SEQ)   // 16384

// ---------------- WMMA types ----------------
typedef __attribute__((ext_vector_type(16))) __bf16       v16bf;
typedef __attribute__((ext_vector_type(8)))  float        v8f;
typedef __attribute__((ext_vector_type(4)))  unsigned int u32x4;
typedef __attribute__((ext_vector_type(8)))  unsigned int u32x8;

union Frag16 { v16bf v; u32x4 q[2]; };

// A-matrix 16x32 bf16 fragment (row-major source, stride in elements).
// Lanes 0-15 row M=lane, halves 0..15 -> K=0..15; lanes 16-31 same M,
// K=16..31 (contiguous, matches B fragment).
__device__ __forceinline__ v16bf load_a_frag_bf16(const __bf16* base, int stride, int lane) {
  const __bf16* p = base + (size_t)(lane & 15) * stride + ((lane >> 4) << 4);
  Frag16 f;
  f.q[0] = *(const u32x4*)(p);        // 8 bf16, K = koff+0..7
  f.q[1] = *(const u32x4*)(p + 8);    // 8 bf16, K = koff+8..15
  return f.v;
}

// B-matrix 32x16 bf16 fragment, memory contiguous along K for a fixed column:
// element(K=kb+i, N=n) = base[n*stride + kb + i].
// Lanes 0-15: N=lane, K=0..15 in halves 0..15; lanes 16-31: K=16..31.
__device__ __forceinline__ v16bf load_b_frag_kcontig(const __bf16* base, long stride, int lane) {
  const __bf16* p = base + (long)(lane & 15) * stride + ((lane >> 4) << 4);
  Frag16 f;
  f.q[0] = *(const u32x4*)(p);        // K = kb+0..7
  f.q[1] = *(const u32x4*)(p + 8);    // K = kb+8..15
  return f.v;
}

// ============== Kernel 1: one projection (fp32 -> bf16, f32 accum) ==============
// grid = (NROWS/16, HD/64), block = 128 (4 waves). Wave w handles cols y*64+w*16.
// TRANSPOSED=false: dst row-major [B*S, H]. TRANSPOSED=true: dst [B, H, S] (V^T).
template <bool TRANSPOSED>
__global__ __launch_bounds__(128) void proj_kernel(
    const float* __restrict__ x, const float* __restrict__ W,
    const float* __restrict__ bias, __bf16* __restrict__ dst)
{
  const int lane = threadIdx.x & 31;
  const int wave = threadIdx.x >> 5;
  const int row0 = blockIdx.x * 16;
  const int col0 = blockIdx.y * 64 + wave * 16;

  const int m    = lane & 15;
  const int n    = lane & 15;
  const int koff = (lane >> 4) << 4;   // A K offset
  const int kb   = (lane >> 4) << 4;   // B K offset

  v8f acc = {};
  for (int k0 = 0; k0 < FD; k0 += 32) {
    v16bf a, b;
    const float* ap = x + (size_t)(row0 + m) * FD + k0 + koff;
    #pragma unroll
    for (int i = 0; i < 8; ++i) { a[i] = (__bf16)ap[i]; a[8 + i] = (__bf16)ap[8 + i]; }
    const float* bp = W + (size_t)(k0 + kb) * HD + col0 + n;
    #pragma unroll
    for (int i = 0; i < 16; ++i) b[i] = (__bf16)bp[(size_t)i * HD];
    acc = __builtin_amdgcn_wmma_f32_16x16x32_bf16(false, a, false, b, (short)0, acc, false, false);
  }

  const float bval = bias[col0 + n];
  const int hi8 = (lane >> 4) << 3;   // C/D: VGPR r -> row r or r+8
  #pragma unroll
  for (int r = 0; r < 8; ++r) {
    const int row = row0 + hi8 + r;
    const __bf16 val = (__bf16)(acc[r] + bval);
    if (TRANSPOSED) {
      const int bb = row >> 11;             // SEQ = 2048 rows per batch
      const int s  = row & (SEQ - 1);
      dst[(size_t)bb * HD * SEQ + (size_t)(col0 + n) * SEQ + s] = val;
    } else {
      dst[(size_t)row * HD + col0 + n] = val;
    }
  }
}

// ============== Kernel 2: flash attention over one (batch, 16-query tile) ==============
// grid = (SEQ/16, BATCH), block = 256 (8 waves).
__global__ __launch_bounds__(256) void attn_kernel(
    const __bf16* __restrict__ qbuf, const __bf16* __restrict__ kbuf,
    const __bf16* __restrict__ vT, float* __restrict__ out)
{
  __shared__ __bf16 Qs[16][HD];        // 16 KB query tile (TDM destination)
  __shared__ float  Outs[16][HD];      // 32 KB fp32 output accumulator
  __shared__ float  Sc[16][128];       //  8 KB raw scores for one superblock
  __shared__ __bf16 Pb[16][128];       //  4 KB exp(scores) in bf16
  __shared__ float  rowm[16], rowl[16], rowscale[16];

  const int t    = threadIdx.x;
  const int lane = t & 31;
  const int wave = t >> 5;
  const int qt   = blockIdx.x;
  const int bidx = blockIdx.y;
  const size_t qkbase = (size_t)(bidx * SEQ + qt * 16) * HD;

  // ---- TDM: DMA the contiguous 16 KB Q tile global->LDS (wave 0 issues) ----
  if (wave == 0) {
    const unsigned long long gaddr =
        (unsigned long long)(const void*)(qbuf + qkbase);
    const unsigned ldsoff =
        (unsigned)(unsigned long long)(__attribute__((address_space(3))) __bf16*)&Qs[0][0];
    // D# group 0: count=1, lds_addr, 57-bit global_addr, type=2 ("image")
    u32x4 g0;
    g0[0] = 1u;                                   // count = 1 (valid descriptor)
    g0[1] = ldsoff;                               // lds_addr
    g0[2] = (unsigned)gaddr;                      // global_addr[31:0]
    g0[3] = (unsigned)((gaddr >> 32) & 0x1FFFFFFu) | 0x80000000u; // addr[56:32] | type=2
    // D# group 1: 1-D tile of 8192 bf16 elements (data_size=2B)
    u32x8 g1;
    g1[0] = 0x00010000u;                          // wg_mask=0, data_size=1 (2 bytes)
    g1[1] = (8192u << 16);                        // tensor_dim0[15:0]
    g1[2] = (1u << 16);                           // tensor_dim0[31:16]=0, tensor_dim1=1
    g1[3] = (8192u << 16);                        // tensor_dim1 hi=0, tile_dim0=8192
    g1[4] = 1u;                                   // tile_dim1=1, tile_dim2=0
    g1[5] = 8192u;                                // tensor_dim0_stride[31:0]
    g1[6] = (8192u << 16);                        // stride0 hi=0, tensor_dim1_stride lo
    g1[7] = 0u;
    asm volatile("tensor_load_to_lds %0, %1\n\t"
                 "s_wait_tensorcnt 0x0"
                 :: "s"(g0), "s"(g1) : "memory");
  }

  // ---- init accumulators while the DMA is in flight elsewhere ----
  {
    float* of = &Outs[0][0];
    for (int i = t; i < 16 * HD; i += 256) of[i] = 0.0f;
    if (t < 16) { rowm[t] = -1e30f; rowl[t] = 0.0f; }
  }
  __syncthreads();

  const float scale = 0.044194173824159216f;  // 1/sqrt(512)
  const int hi8 = (lane >> 4) << 3;
  const int n   = lane & 15;
  const int row = t >> 4;   // softmax: 16 contiguous lanes per row (same wave)
  const int seg = t & 15;

  for (int sb = 0; sb < SEQ / 128; ++sb) {
    // ---- Phase A: scores for this wave's 16 keys, K-loop over HD ----
    {
      const int key0 = sb * 128 + wave * 16;
      const __bf16* kb_base = kbuf + (size_t)(bidx * SEQ + key0) * HD;
      const __bf16* pfp = kb_base + (size_t)(lane & 15) * HD + ((lane >> 4) << 4);
      v8f acc = {};
      for (int k0 = 0; k0 < HD; k0 += 128) {
        __builtin_prefetch((const void*)(pfp + k0 + 128), 0, 0);  // next K chunk
        #pragma unroll
        for (int kk = 0; kk < 4; ++kk) {
          const int k = k0 + kk * 32;
          v16bf a = load_a_frag_bf16(&Qs[0][k], HD, lane);        // Q[16, k..k+32)
          v16bf b = load_b_frag_kcontig(kb_base + k, HD, lane);   // K^T[k.., 16 keys]
          acc = __builtin_amdgcn_wmma_f32_16x16x32_bf16(false, a, false, b,
                                                        (short)0, acc, false, false);
        }
      }
      #pragma unroll
      for (int r = 0; r < 8; ++r)
        Sc[hi8 + r][wave * 16 + n] = acc[r] * scale;
    }
    __syncthreads();

    // ---- Phase B: online softmax update (16 threads per row) ----
    {
      float mx = -1e30f;
      #pragma unroll
      for (int i = 0; i < 8; ++i) mx = fmaxf(mx, Sc[row][seg * 8 + i]);
      #pragma unroll
      for (int off = 8; off >= 1; off >>= 1) mx = fmaxf(mx, __shfl_xor(mx, off, 16));
      const float m_old = rowm[row];
      const float m_new = fmaxf(m_old, mx);
      float sum = 0.0f;
      #pragma unroll
      for (int i = 0; i < 8; ++i) {
        const float p = __expf(Sc[row][seg * 8 + i] - m_new);
        Pb[row][seg * 8 + i] = (__bf16)p;
        sum += p;
      }
      #pragma unroll
      for (int off = 8; off >= 1; off >>= 1) sum += __shfl_xor(sum, off, 16);
      if (seg == 0) {
        const float r = __expf(m_old - m_new);
        rowm[row] = m_new;
        rowl[row] = rowl[row] * r + sum;
        rowscale[row] = r;
      }
    }
    __syncthreads();

    // ---- rescale accumulated output by exp(m_old - m_new) ----
    {
      float* of = &Outs[0][0];
      for (int i = t; i < 16 * HD; i += 256) of[i] *= rowscale[i >> 9];
    }
    __syncthreads();

    // ---- Phase C: PV — wave owns output columns [wave*64, wave*64+64) ----
    {
      const size_t vbase0 = (size_t)bidx * HD * SEQ + (size_t)sb * 128;
      #pragma unroll
      for (int ht = 0; ht < 4; ++ht) {
        const int col0 = wave * 64 + ht * 16;
        v8f c = {};
        #pragma unroll
        for (int kk = 0; kk < 4; ++kk) {
          v16bf a = load_a_frag_bf16(&Pb[0][kk * 32], 128, lane);  // P[16, 32 keys]
          v16bf b = load_b_frag_kcontig(vT + vbase0 + (size_t)col0 * SEQ + kk * 32,
                                        SEQ, lane);
          c = __builtin_amdgcn_wmma_f32_16x16x32_bf16(false, a, false, b,
                                                      (short)0, c, false, false);
        }
        #pragma unroll
        for (int r = 0; r < 8; ++r)
          Outs[hi8 + r][col0 + n] += c[r];
      }
    }
    __syncthreads();
  }

  // ---- finalize: divide by softmax denominator, store fp32 ----
  for (int i = t; i < 16 * HD; i += 256) {
    const int rr = i >> 9, cc = i & (HD - 1);
    out[qkbase + (size_t)rr * HD + cc] = Outs[rr][cc] / rowl[rr];
  }
}

// ============== launch ==============
extern "C" void kernel_launch(void* const* d_in, const int* in_sizes, int n_in,
                              void* d_out, int out_size, void* d_ws, size_t ws_size,
                              hipStream_t stream) {
  (void)in_sizes; (void)n_in; (void)out_size; (void)ws_size;
  const float* x  = (const float*)d_in[0];
  const float* Wq = (const float*)d_in[1];
  const float* bq = (const float*)d_in[2];
  const float* Wk = (const float*)d_in[3];
  const float* bk = (const float*)d_in[4];
  const float* Wv = (const float*)d_in[5];
  const float* bv = (const float*)d_in[6];
  float* out = (float*)d_out;

  const size_t per = (size_t)NROWS * HD;   // 8M bf16 elements each
  __bf16* qbuf = (__bf16*)d_ws;            // [B*S, H]
  __bf16* kbuf = qbuf + per;               // [B*S, H]
  __bf16* vT   = kbuf + per;               // [B, H, S]

  const dim3 pgrid(NROWS / 16, HD / 64);
  proj_kernel<false><<<pgrid, 128, 0, stream>>>(x, Wq, bq, qbuf);
  proj_kernel<false><<<pgrid, 128, 0, stream>>>(x, Wk, bk, kbuf);
  proj_kernel<true ><<<pgrid, 128, 0, stream>>>(x, Wv, bv, vT);
  attn_kernel<<<dim3(SEQ / 16, BATCH), 256, 0, stream>>>(qbuf, kbuf, vT, out);
}
